// PyrDCNAlign_14740327759935
// MI455X (gfx1250) — matrix operand order repaired and checked by reading, not compiled, not run on hardware
//
#include <hip/hip_runtime.h>

typedef __attribute__((ext_vector_type(16))) _Float16 v16h;
typedef __attribute__((ext_vector_type(8)))  float    v8f;

#define DEV __device__ __forceinline__

// A-fragment K offset within a 32-wide chunk for 16-bit WMMA A (16x32): contiguous k = 16*hf + e (matches B fills).
// One contiguous 16-half run per lane half.
DEV int kmapA(int e, int hf) { return e + hf * 16; }

// PyTorch bicubic kernel, a = -0.75
DEV float cubw(float t) {
  float at = fabsf(t);
  if (at <= 1.f) return (1.25f * at - 2.25f) * at * at + 1.f;
  if (at < 2.f)  return ((-0.75f * at + 3.75f) * at - 6.f) * at + 3.f;
  return 0.f;
}

// ---------------- bicubic resize (align_corners=False, border clamp) ----------------
// Fully generalized strides: source/dest may live in concatenated and/or zero-halo
// padded [B][C][Hp][Wp] buffers (CS=channel stride, RS=row stride, Off=origin offset).
__global__ void k_resize_bicubic(const float* __restrict__ in, long inBS, long inCS,
                                 int inRS, int inOff, int cBaseIn,
                                 float* __restrict__ out, long outBS, long outCS,
                                 int outRS, int outOff, int cBaseOut,
                                 int Bn, int Cch, int Hi, int Wi, int Ho, int Wo, float rs) {
  long i = (long)blockIdx.x * blockDim.x + threadIdx.x;
  long tot = (long)Bn * Cch * Ho * Wo;
  if (i >= tot) return;
  int ox = (int)(i % Wo);
  int oy = (int)((i / Wo) % Ho);
  long t2 = i / ((long)Wo * Ho);
  int cc = (int)(t2 % Cch);
  int b  = (int)(t2 / Cch);
  float sy = (oy + 0.5f) * rs - 0.5f;
  float sx = (ox + 0.5f) * rs - 0.5f;
  int fy = (int)floorf(sy), fx = (int)floorf(sx);
  float ty = sy - (float)fy, tx = sx - (float)fx;
  float wy[4], wx[4];
  int iy[4], ix[4];
#pragma unroll
  for (int o = 0; o < 4; ++o) {
    wy[o] = cubw(ty - (float)(o - 1));
    wx[o] = cubw(tx - (float)(o - 1));
    iy[o] = min(max(fy + o - 1, 0), Hi - 1);
    ix[o] = min(max(fx + o - 1, 0), Wi - 1);
  }
  const float* src = in + (long)b * inBS + (long)(cBaseIn + cc) * inCS + inOff;
  float acc = 0.f;
#pragma unroll
  for (int y = 0; y < 4; ++y) {
    float rowacc = 0.f;
#pragma unroll
    for (int x = 0; x < 4; ++x) rowacc += wx[x] * src[(long)iy[y] * inRS + ix[x]];
    acc += wy[y] * rowacc;
  }
  out[(long)b * outBS + (long)(cBaseOut + cc) * outCS + (long)oy * outRS + ox + outOff] = acc;
}

// copy [B][C][H][W] -> zero-halo padded [B][*][H+4][W+4] at channel base cBaseOut
__global__ void k_pad_copy(const float* __restrict__ src, long srcBS,
                           float* __restrict__ dst, long dstBS, int cBaseOut,
                           int Bn, int Cch, int H, int W) {
  int Hp = H + 4, Wp = W + 4;
  long i = (long)blockIdx.x * blockDim.x + threadIdx.x;
  long tot = (long)Bn * Cch * Hp * Wp;
  if (i >= tot) return;
  int xp = (int)(i % Wp);
  int yp = (int)((i / Wp) % Hp);
  long t2 = i / ((long)Wp * Hp);
  int cc = (int)(t2 % Cch);
  int b  = (int)(t2 / Cch);
  int y = yp - 2, x = xp - 2;
  float v = ((unsigned)y < (unsigned)H && (unsigned)x < (unsigned)W)
              ? src[(long)b * srcBS + (long)cc * H * W + (long)y * W + x] : 0.f;
  dst[(long)b * dstBS + (long)(cBaseOut + cc) * Hp * Wp + (long)yp * Wp + xp] = v;
}

__global__ void k_fill0(float* __restrict__ p, long n) {
  long i = (long)blockIdx.x * blockDim.x + threadIdx.x;
  if (i < n) p[i] = 0.f;
}

// zero a channel range of a strided [B][*][P] tensor (pad channels of concat buffers)
__global__ void k_zero_ch(float* __restrict__ dst, long bs, int cBase, int nch, int P, int Bn) {
  long i = (long)blockIdx.x * blockDim.x + threadIdx.x;
  long tot = (long)Bn * nch * P;
  if (i >= tot) return;
  int pp = (int)(i % P);
  long t2 = i / P;
  int cc = (int)(t2 % nch);
  int b  = (int)(t2 / nch);
  dst[(long)b * bs + (long)(cBase + cc) * P + pp] = 0.f;
}

// im2col byte-offset table for 5x5 conv over a padded image: k -> 4*(c*Pp + ky*Wp + kx)
__global__ void k_build_tab(int* __restrict__ tab, int n, int Pp, int Wp) {
  int k = blockIdx.x * blockDim.x + threadIdx.x;
  if (k >= n) return;
  int c = k / 25, tap = k - c * 25;
  tab[k] = 4 * (c * Pp + (tap / 5) * Wp + (tap % 5));
}

// ---------------- weight packing to f16 ----------------
__global__ void k_pack_direct(const float* __restrict__ w, _Float16* __restrict__ dst,
                              int M, int K, int Mpad, int Kpad) {
  int i = blockIdx.x * blockDim.x + threadIdx.x;
  if (i >= Mpad * Kpad) return;
  int k = i % Kpad, m = i / Kpad;
  dst[i] = (m < M && k < K) ? (_Float16)w[(long)m * K + k] : (_Float16)0.f;
}

// deform weights: w (O=C, C, 3,3) -> A[o][t*C + c] (tap-major K so a 32-chunk shares one sample set)
__global__ void k_pack_deform(const float* __restrict__ w, _Float16* __restrict__ dst, int C) {
  int i = blockIdx.x * blockDim.x + threadIdx.x;
  if (i >= C * C * 9) return;
  int c = i % C;
  int t = (i / C) % 9;
  int o = i / (9 * C);
  dst[i] = (_Float16)w[((long)o * C + c) * 9 + t];
}

// ---------------- 5x5 conv via WMMA, zero-halo padded single input ----------------
// Input: padded [B][Ctot][H+4][W+4] with zero halo -> every load is in-bounds, no branches.
// Output channel stride == P (compile-time) -> immediate-offset stores.
template <int MT, int NT, int H, int W, int COUT>
__global__ void k_conv5_wmma(const float* __restrict__ in, long inBS,
                             const int* __restrict__ offTab,
                             const _Float16* __restrict__ Af,
                             const float* __restrict__ bias,
                             float* __restrict__ out, long outBS,
                             int K, int tilesPerBatch, int numTiles) {
  constexpr int Wp = W + 4, P = H * W;
  int wave = (int)((blockIdx.x * blockDim.x + threadIdx.x) >> 5);
  int lane = threadIdx.x & 31;
  if (wave >= numTiles) return;
  int b  = wave / tilesPerBatch;
  int p0 = (wave - b * tilesPerBatch) * (16 * NT);
  int n = lane & 15, hf = lane >> 4;
  int p = p0 + n;
  int py = p / W, px = p - py * W;
  // tap (ky,kx) of output pixel reads padded element (py+ky)*Wp + (px+kx)
  const char* srcb = (const char*)(in + (long)b * inBS + (long)py * Wp + px);
  v8f acc[MT][NT] = {};
  for (int k0 = 0; k0 < K; k0 += 32) {
    int kb = k0 + hf * 16;
    const int* tb = offTab + kb;
    v16h bf[NT];
#pragma unroll
    for (int e = 0; e < 16; ++e) {
      const float* s = (const float*)(srcb + tb[e]);
#pragma unroll
      for (int j = 0; j < NT; ++j) bf[j][e] = (_Float16)s[j * 16];
    }
#pragma unroll
    for (int t = 0; t < MT; ++t) {
      v16h af;
      const _Float16* arow = Af + (long)(t * 16 + n) * K + k0;
#pragma unroll
      for (int e = 0; e < 16; ++e) af[e] = arow[kmapA(e, hf)];
#pragma unroll
      for (int j = 0; j < NT; ++j)
        acc[t][j] = __builtin_amdgcn_wmma_f32_16x16x32_f16(false, af, false, bf[j],
                                                           (short)0, acc[t][j], false, false);
    }
  }
  float* ob = out + (long)b * outBS + p + (long)(hf * 8) * P;   // half-wave row base
  const float* bi = bias + hf * 8;
#pragma unroll
  for (int t = 0; t < MT; ++t) {
#pragma unroll
    for (int r = 0; r < 8; ++r) {
      if (t * 16 + r + hf * 8 < COUT) {   // statically resolved except the COUT=27 corner
#pragma unroll
        for (int j = 0; j < NT; ++j)
          ob[(long)(t * 16 + r) * P + j * 16] = acc[t][j][r] + bi[t * 16 + r];
      }
    }
  }
}

// ---------------- 1x1 conv via WMMA over a pre-concatenated (unpadded) input ----------------
// OCS = output channel stride (compile-time; != P only for the padded a1 write).
template <int MT, int NT, int H, int W, int OCS, int COUT>
__global__ void k_conv1_wmma(const float* __restrict__ in, long inBS,
                             const _Float16* __restrict__ Af,
                             const float* __restrict__ bias,
                             float* __restrict__ out, long outBS,
                             int outRS, int outOff,
                             int Kpad, int tilesPerBatch, int numTiles) {
  constexpr int P = H * W;
  int wave = (int)((blockIdx.x * blockDim.x + threadIdx.x) >> 5);
  int lane = threadIdx.x & 31;
  if (wave >= numTiles) return;
  int b  = wave / tilesPerBatch;
  int p0 = (wave - b * tilesPerBatch) * (16 * NT);
  int n = lane & 15, hf = lane >> 4;
  int p = p0 + n;
  int py = p / W, px = p - py * W;
  v8f acc[MT][NT] = {};
  for (int k0 = 0; k0 < Kpad; k0 += 32) {
    int kb = k0 + hf * 16;
    const float* src = in + (long)b * inBS + (long)kb * P + p;
    v16h bf[NT];
#pragma unroll
    for (int e = 0; e < 16; ++e)
#pragma unroll
      for (int j = 0; j < NT; ++j)
        bf[j][e] = (_Float16)src[(long)e * P + j * 16];
#pragma unroll
    for (int t = 0; t < MT; ++t) {
      v16h af;
      const _Float16* arow = Af + (long)(t * 16 + n) * Kpad + k0;
#pragma unroll
      for (int e = 0; e < 16; ++e) af[e] = arow[kmapA(e, hf)];
#pragma unroll
      for (int j = 0; j < NT; ++j)
        acc[t][j] = __builtin_amdgcn_wmma_f32_16x16x32_f16(false, af, false, bf[j],
                                                           (short)0, acc[t][j], false, false);
    }
  }
  float* ob = out + (long)b * outBS + (long)py * outRS + px + outOff + (long)(hf * 8) * OCS;
  const float* bi = bias + hf * 8;
#pragma unroll
  for (int t = 0; t < MT; ++t) {
#pragma unroll
    for (int r = 0; r < 8; ++r) {
      if (t * 16 + r + hf * 8 < COUT) {
#pragma unroll
        for (int j = 0; j < NT; ++j)
          ob[(long)(t * 16 + r) * OCS + j * 16] = acc[t][j][r] + bi[t * 16 + r];
      }
    }
  }
}

// ---------------- modulated deformable 3x3 conv (DCNv2) via WMMA ----------------
// X in zero-halo padded layout [*][H+4][W+4]; clamped bilinear only touches the interior.
// 4 corner base pointers + immediate offsets e*Pp; OM/out unpadded.
template <int H, int W>
__global__ void k_dcn_wmma(const float* __restrict__ X, long xBS,
                           const float* __restrict__ OM, long omBS,
                           const _Float16* __restrict__ Af,
                           const float* __restrict__ bias,
                           float* __restrict__ out, long outBS,
                           int tilesPerBatch, int numTiles) {
  constexpr int P = H * W, Wp = W + 4, Pp = (H + 4) * (W + 4);
  const int C = 64;
  int wave = (int)((blockIdx.x * blockDim.x + threadIdx.x) >> 5);
  int lane = threadIdx.x & 31;
  if (wave >= numTiles) return;
  int b  = wave / tilesPerBatch;
  int p0 = (wave - b * tilesPerBatch) * 16;
  int n = lane & 15, hf = lane >> 4;
  int p = p0 + n;
  int py = p / W, px = p - py * W;
  const float* xbase = X + (long)b * xBS + (long)(hf * 16) * Pp + (2 * Wp + 2);
  const float* omp = OM + (long)b * omBS;
  v8f acc[4] = {};
#pragma unroll 1
  for (int t = 0; t < 9; ++t) {
    // sampling parameters for (tap t, pixel p) -- shared across all 64 channels
    float dy = omp[(long)(2 * t) * P + p];
    float dx = omp[(long)(2 * t + 1) * P + p];
    float mk = 1.f / (1.f + __expf(-omp[(long)(18 + t) * P + p]));
    float sy = (float)py + (float)(t / 3 - 1) + dy;
    float sx = (float)px + (float)(t % 3 - 1) + dx;
    float fy = floorf(sy), fx = floorf(sx);
    float ly = sy - fy, lx = sx - fx;
    int y0 = (int)fy, x0 = (int)fx;
    int y1 = y0 + 1, x1 = x0 + 1;
    float w00 = (1.f - ly) * (1.f - lx), w01 = (1.f - ly) * lx;
    float w10 = ly * (1.f - lx),         w11 = ly * lx;
    bool vy0 = (unsigned)y0 < (unsigned)H, vy1 = (unsigned)y1 < (unsigned)H;
    bool vx0 = (unsigned)x0 < (unsigned)W, vx1 = (unsigned)x1 < (unsigned)W;
    if (!(vy0 && vx0)) w00 = 0.f;
    if (!(vy0 && vx1)) w01 = 0.f;
    if (!(vy1 && vx0)) w10 = 0.f;
    if (!(vy1 && vx1)) w11 = 0.f;
    w00 *= mk; w01 *= mk; w10 *= mk; w11 *= mk;
    int cy0 = min(max(y0, 0), H - 1), cy1 = min(max(y1, 0), H - 1);
    int cx0 = min(max(x0, 0), W - 1), cx1 = min(max(x1, 0), W - 1);
    int i00 = cy0 * Wp + cx0, i01 = cy0 * Wp + cx1;
    int i10 = cy1 * Wp + cx0, i11 = cy1 * Wp + cx1;
#pragma unroll
    for (int c0 = 0; c0 < C; c0 += 32) {
      int k0 = t * C + c0;
      const float* q = xbase + (long)c0 * Pp;
      const float* q00 = q + i00;
      const float* q01 = q + i01;
      const float* q10 = q + i10;
      const float* q11 = q + i11;
      v16h bf;
#pragma unroll
      for (int e = 0; e < 16; ++e) {
        float v = w00 * q00[(long)e * Pp] + w01 * q01[(long)e * Pp]
                + w10 * q10[(long)e * Pp] + w11 * q11[(long)e * Pp];
        bf[e] = (_Float16)v;
      }
#pragma unroll
      for (int tt = 0; tt < 4; ++tt) {
        v16h af;
        const _Float16* arow = Af + (long)(tt * 16 + n) * 576 + k0;
#pragma unroll
        for (int e = 0; e < 16; ++e) af[e] = arow[kmapA(e, hf)];
        acc[tt] = __builtin_amdgcn_wmma_f32_16x16x32_f16(false, af, false, bf,
                                                         (short)0, acc[tt], false, false);
      }
    }
  }
  float* ob = out + (long)b * outBS + p + (long)(hf * 8) * P;
  const float* bi = bias + hf * 8;
#pragma unroll
  for (int tt = 0; tt < 4; ++tt) {
#pragma unroll
    for (int r = 0; r < 8; ++r)
      ob[(long)(tt * 16 + r) * P] = acc[tt][r] + bi[tt * 16 + r];
  }
}

// ---------------- host orchestration ----------------
static inline size_t align256(size_t x) { return (x + 255) & ~(size_t)255; }
static inline int cdiv(long a, long b) { return (int)((a + b - 1) / b); }

extern "C" void kernel_launch(void* const* d_in, const int* in_sizes, int n_in,
                              void* d_out, int out_size, void* d_ws, size_t ws_size,
                              hipStream_t stream) {
  (void)in_sizes; (void)n_in; (void)out_size; (void)ws_size;
  const int B = 8, C = 64;
  const int H1 = 128, W1 = 128, P1 = H1 * W1;
  const int H2 = 64,  W2 = 64,  P2 = H2 * W2;
  const int H3 = 32,  W3 = 32,  P3 = H3 * W3;
  const int Wp1 = W1 + 4, Pp1 = (H1 + 4) * Wp1;   // 132*132 = 17424
  const int Wp2 = W2 + 4, Pp2 = (H2 + 4) * Wp2;
  const int Wp3 = W3 + 4, Pp3 = (H3 + 4) * Wp3;
  const long ACSZ = (long)B * C * P1;

  const float* feat_c = (const float*)d_in[0];
  const float* feat_n = (const float*)d_in[1];
  const float* w_om1 = (const float*)d_in[2];  const float* b_om1 = (const float*)d_in[3];
  const float* w_om2 = (const float*)d_in[4];  const float* b_om2 = (const float*)d_in[5];
  const float* w_om3 = (const float*)d_in[6];  const float* b_om3 = (const float*)d_in[7];
  const float* w_omc = (const float*)d_in[8];  const float* b_omc = (const float*)d_in[9];
  const float* w_f23 = (const float*)d_in[10]; const float* b_f23 = (const float*)d_in[11];
  const float* w_f12 = (const float*)d_in[12]; const float* b_f12 = (const float*)d_in[13];
  const float* w_a23 = (const float*)d_in[14]; const float* b_a23 = (const float*)d_in[15];
  const float* w_a12 = (const float*)d_in[16]; const float* b_a12 = (const float*)d_in[17];
  const float* w_d1  = (const float*)d_in[18]; const float* b_d1  = (const float*)d_in[19];
  const float* w_d2  = (const float*)d_in[20]; const float* b_d2  = (const float*)d_in[21];
  const float* w_d3  = (const float*)d_in[22]; const float* b_d3  = (const float*)d_in[23];
  const float* w_dc  = (const float*)d_in[24]; const float* b_dc  = (const float*)d_in[25];

  float* out = (float*)d_out;
  float* out_om1 = out + ACSZ;               // [B][54][P1], om1 at channels 0..26
  float* out_omc = out + ACSZ + (long)27 * P1;

  char* ws = (char*)d_ws;
  size_t cur = 0;
  auto allocF = [&](size_t nel) { float* q = (float*)(ws + cur); cur += align256(nel * 4); return q; };
  auto allocH = [&](size_t nel) { _Float16* q = (_Float16*)(ws + cur); cur += align256(nel * 2); return q; };
  auto allocI = [&](size_t nel) { int* q = (int*)(ws + cur); cur += align256(nel * 4); return q; };

  float* pad_f1  = allocF((size_t)B * 128 * Pp1); // padded: feat_c(0..63) | feat_n -> later a1 (64..127)
  float* pyr2    = allocF((size_t)B * 128 * Pp2); // padded: fc2 | fn2
  float* pyr3    = allocF((size_t)B * 128 * Pp3); // padded: fc3 | fn3
  float* om3     = allocF((size_t)B * 27 * P3);
  float* cat_om2 = allocF((size_t)B * 64 * P2);   // om2_raw(0..26) | om3up(27..53) | pad(54..63)=0
  float* om2     = allocF((size_t)B * 27 * P2);
  float* cat_om1 = allocF((size_t)B * 64 * P1);   // om1_raw | om2up | pad=0
  float* a3      = allocF((size_t)B * 64 * P3);
  float* cat_a2  = allocF((size_t)B * 128 * P2);  // a2_raw(0..63) | a3up(64..127)
  float* a2      = allocF((size_t)B * 64 * P2);
  float* cat_a1  = allocF((size_t)B * 128 * P1);  // a1_raw | a2up
  _Float16* A_om1 = allocH((size_t)32 * 3200); _Float16* A_om2 = allocH((size_t)32 * 3200);
  _Float16* A_om3 = allocH((size_t)32 * 3200); _Float16* A_omc = allocH((size_t)32 * 3200);
  _Float16* A_f23 = allocH((size_t)32 * 64);   _Float16* A_f12 = allocH((size_t)32 * 64);
  _Float16* A_a23 = allocH((size_t)64 * 128);  _Float16* A_a12 = allocH((size_t)64 * 128);
  _Float16* A_d1  = allocH((size_t)64 * 576);  _Float16* A_d2  = allocH((size_t)64 * 576);
  _Float16* A_d3  = allocH((size_t)64 * 576);  _Float16* A_dc  = allocH((size_t)64 * 576);
  int* tab1 = allocI(3200); int* tab2 = allocI(3200); int* tab3 = allocI(3200);

  const int T = 256;
  // im2col offset tables + packed weights
  k_build_tab<<<cdiv(3200, T), T, 0, stream>>>(tab1, 3200, Pp1, Wp1);
  k_build_tab<<<cdiv(3200, T), T, 0, stream>>>(tab2, 3200, Pp2, Wp2);
  k_build_tab<<<cdiv(3200, T), T, 0, stream>>>(tab3, 3200, Pp3, Wp3);
  k_pack_direct<<<cdiv(32 * 3200, T), T, 0, stream>>>(w_om1, A_om1, 27, 3200, 32, 3200);
  k_pack_direct<<<cdiv(32 * 3200, T), T, 0, stream>>>(w_om2, A_om2, 27, 3200, 32, 3200);
  k_pack_direct<<<cdiv(32 * 3200, T), T, 0, stream>>>(w_om3, A_om3, 27, 3200, 32, 3200);
  k_pack_direct<<<cdiv(32 * 3200, T), T, 0, stream>>>(w_omc, A_omc, 27, 3200, 32, 3200);
  k_pack_direct<<<cdiv(32 * 64, T), T, 0, stream>>>(w_f23, A_f23, 27, 54, 32, 64);
  k_pack_direct<<<cdiv(32 * 64, T), T, 0, stream>>>(w_f12, A_f12, 27, 54, 32, 64);
  k_pack_direct<<<cdiv(64 * 128, T), T, 0, stream>>>(w_a23, A_a23, 64, 128, 64, 128);
  k_pack_direct<<<cdiv(64 * 128, T), T, 0, stream>>>(w_a12, A_a12, 64, 128, 64, 128);
  k_pack_deform<<<cdiv(64 * 576, T), T, 0, stream>>>(w_d1, A_d1, 64);
  k_pack_deform<<<cdiv(64 * 576, T), T, 0, stream>>>(w_d2, A_d2, 64);
  k_pack_deform<<<cdiv(64 * 576, T), T, 0, stream>>>(w_d3, A_d3, 64);
  k_pack_deform<<<cdiv(64 * 576, T), T, 0, stream>>>(w_dc, A_dc, 64);

  // zero halos (pyr buffers) and pad channels of 54->64 concat buffers
  k_fill0<<<cdiv((long)B * 128 * Pp2, T), T, 0, stream>>>(pyr2, (long)B * 128 * Pp2);
  k_fill0<<<cdiv((long)B * 128 * Pp3, T), T, 0, stream>>>(pyr3, (long)B * 128 * Pp3);
  k_zero_ch<<<cdiv((long)B * 10 * P2, T), T, 0, stream>>>(cat_om2, (long)64 * P2, 54, 10, P2, B);
  k_zero_ch<<<cdiv((long)B * 10 * P1, T), T, 0, stream>>>(cat_om1, (long)64 * P1, 54, 10, P1, B);

  // level-1 padded concat: feat_c | feat_n  (feat_n slot is later overwritten by a1)
  k_pad_copy<<<cdiv((long)B * C * Pp1, T), T, 0, stream>>>(feat_c, (long)64 * P1,
      pad_f1, (long)128 * Pp1, 0, B, C, H1, W1);
  k_pad_copy<<<cdiv((long)B * C * Pp1, T), T, 0, stream>>>(feat_n, (long)64 * P1,
      pad_f1, (long)128 * Pp1, 64, B, C, H1, W1);

  // pyramid (rs = Hin/Hout) into padded layouts
  const int pOff2 = 2 * Wp2 + 2, pOff3 = 2 * Wp3 + 2, pOff1 = 2 * Wp1 + 2;
  k_resize_bicubic<<<cdiv((long)B * C * P2, T), T, 0, stream>>>(feat_c, (long)64 * P1, P1, W1, 0, 0,
      pyr2, (long)128 * Pp2, Pp2, Wp2, pOff2, 0, B, C, H1, W1, H2, W2, 2.0f);
  k_resize_bicubic<<<cdiv((long)B * C * P2, T), T, 0, stream>>>(feat_n, (long)64 * P1, P1, W1, 0, 0,
      pyr2, (long)128 * Pp2, Pp2, Wp2, pOff2, 64, B, C, H1, W1, H2, W2, 2.0f);
  k_resize_bicubic<<<cdiv((long)B * C * P3, T), T, 0, stream>>>(pyr2, (long)128 * Pp2, Pp2, Wp2, pOff2, 0,
      pyr3, (long)128 * Pp3, Pp3, Wp3, pOff3, 0, B, C, H2, W2, H3, W3, 2.0f);
  k_resize_bicubic<<<cdiv((long)B * C * P3, T), T, 0, stream>>>(pyr2, (long)128 * Pp2, Pp2, Wp2, pOff2, 64,
      pyr3, (long)128 * Pp3, Pp3, Wp3, pOff3, 64, B, C, H2, W2, H3, W3, 2.0f);

  auto cB = [&](int P) { return cdiv((long)B * (P / 32), 8); };   // conv: 32 px/wave, 8 waves/block
  auto dB = [&](int P) { return cdiv((long)B * (P / 16), 8); };   // dcn: 16 px/wave

  // offset/mask branch
  k_conv5_wmma<2, 2, 32, 32, 27><<<cB(P3), T, 0, stream>>>(pyr3, (long)128 * Pp3, tab3,
      A_om3, b_om3, om3, (long)27 * P3, 3200, P3 / 32, B * P3 / 32);
  k_conv5_wmma<2, 2, 64, 64, 27><<<cB(P2), T, 0, stream>>>(pyr2, (long)128 * Pp2, tab2,
      A_om2, b_om2, cat_om2, (long)64 * P2, 3200, P2 / 32, B * P2 / 32);
  k_resize_bicubic<<<cdiv((long)B * 27 * P2, T), T, 0, stream>>>(om3, (long)27 * P3, P3, W3, 0, 0,
      cat_om2, (long)64 * P2, P2, W2, 0, 27, B, 27, H3, W3, H2, W2, 0.5f);
  k_conv1_wmma<2, 2, 64, 64, 4096, 27><<<cB(P2), T, 0, stream>>>(cat_om2, (long)64 * P2,
      A_f23, b_f23, om2, (long)27 * P2, W2, 0, 64, P2 / 32, B * P2 / 32);
  k_conv5_wmma<2, 2, 128, 128, 27><<<cB(P1), T, 0, stream>>>(pad_f1, (long)128 * Pp1, tab1,
      A_om1, b_om1, cat_om1, (long)64 * P1, 3200, P1 / 32, B * P1 / 32);
  k_resize_bicubic<<<cdiv((long)B * 27 * P1, T), T, 0, stream>>>(om2, (long)27 * P2, P2, W2, 0, 0,
      cat_om1, (long)64 * P1, P1, W1, 0, 27, B, 27, H2, W2, H1, W1, 0.5f);
  k_conv1_wmma<2, 2, 128, 128, 16384, 27><<<cB(P1), T, 0, stream>>>(cat_om1, (long)64 * P1,
      A_f12, b_f12, out_om1, (long)54 * P1, W1, 0, 64, P1 / 32, B * P1 / 32);

  // aligned-feature branch
  k_dcn_wmma<32, 32><<<dB(P3), T, 0, stream>>>(pyr3 + (long)64 * Pp3, (long)128 * Pp3,
      om3, (long)27 * P3, A_d3, b_d3, a3, (long)64 * P3, P3 / 16, B * P3 / 16);
  k_dcn_wmma<64, 64><<<dB(P2), T, 0, stream>>>(pyr2 + (long)64 * Pp2, (long)128 * Pp2,
      om2, (long)27 * P2, A_d2, b_d2, cat_a2, (long)128 * P2, P2 / 16, B * P2 / 16);
  k_resize_bicubic<<<cdiv((long)B * C * P2, T), T, 0, stream>>>(a3, (long)64 * P3, P3, W3, 0, 0,
      cat_a2, (long)128 * P2, P2, W2, 0, 64, B, C, H3, W3, H2, W2, 0.5f);
  k_conv1_wmma<4, 2, 64, 64, 4096, 64><<<cB(P2), T, 0, stream>>>(cat_a2, (long)128 * P2,
      A_a23, b_a23, a2, (long)64 * P2, W2, 0, 128, P2 / 32, B * P2 / 32);
  k_dcn_wmma<128, 128><<<dB(P1), T, 0, stream>>>(pad_f1 + (long)64 * Pp1, (long)128 * Pp1,
      out_om1, (long)54 * P1, A_d1, b_d1, cat_a1, (long)128 * P1, P1 / 16, B * P1 / 16);
  k_resize_bicubic<<<cdiv((long)B * C * P1, T), T, 0, stream>>>(a2, (long)64 * P2, P2, W2, 0, 0,
      cat_a1, (long)128 * P1, P1, W1, 0, 64, B, C, H2, W2, H1, W1, 0.5f);
  // a1 overwrites feat_n's padded slot (halo stays zero; interior fully rewritten)
  k_conv1_wmma<4, 2, 128, 128, 17424, 64><<<cB(P1), T, 0, stream>>>(cat_a1, (long)128 * P1,
      A_a12, b_a12, pad_f1 + (long)64 * Pp1, (long)128 * Pp1, Wp1, pOff1,
      128, P1 / 32, B * P1 / 32);

  // cascade refinement
  k_conv5_wmma<2, 2, 128, 128, 27><<<cB(P1), T, 0, stream>>>(pad_f1, (long)128 * Pp1, tab1,
      A_omc, b_omc, out_omc, (long)54 * P1, 3200, P1 / 32, B * P1 / 32);
  k_dcn_wmma<128, 128><<<dB(P1), T, 0, stream>>>(pad_f1 + (long)64 * Pp1, (long)128 * Pp1,
      out_omc, (long)54 * P1, A_dc, b_dc, out, (long)64 * P1, P1 / 16, B * P1 / 16);
}
